// PosAggregator_17635135717524
// MI455X (gfx1250) — matrix operand order repaired and checked by reading, not compiled, not run on hardware
//
#include <hip/hip_runtime.h>

typedef __attribute__((ext_vector_type(16))) _Float16 v16h;
typedef __attribute__((ext_vector_type(8)))  _Float16 v8h;
typedef __attribute__((ext_vector_type(8)))  float    v8f;

#define DIM    256
#define KDIM   512
#define N_TGT  8192
#define E_EDGE 409600
#define WQ_PITCH 520   // halves per row (512 + 8 pad -> conflict-friendly ds_load_b128)

__device__ __forceinline__ void flush_atomic(float* p, const float* acc) {
#pragma unroll
    for (int j = 0; j < 8; ++j)
        unsafeAtomicAdd(p + j, acc[j]);   // global_atomic_add_f32, no return
}

__global__ void __launch_bounds__(256)
fused_edge_kernel(const float* __restrict__ hv, const float* __restrict__ hp,
                  const float* __restrict__ ht, const int* __restrict__ dst,
                  const float* __restrict__ wq, float* __restrict__ outTgt,
                  int numTiles) {
    extern __shared__ char smem[];
    _Float16* wqs = (_Float16*)smem;

    // ---- Stage Wq (f32 global -> f16 LDS, row-major with padded pitch) ----
    const int tid = threadIdx.x;
    for (int i = tid; i < (DIM * KDIM) / 4; i += blockDim.x) {
        float4 v = ((const float4*)wq)[i];
        int idx = i * 4;
        int c = idx >> 9;          // output column (row of Wq)
        int kk = idx & 511;        // K index
        _Float16* p = wqs + c * WQ_PITCH + kk;
        p[0] = (_Float16)v.x; p[1] = (_Float16)v.y;
        p[2] = (_Float16)v.z; p[3] = (_Float16)v.w;
    }
    __syncthreads();

    const int lane  = tid & 31;
    const int mrow  = lane & 15;   // M (A rows) / N (C cols)
    const int khi   = lane >> 4;   // half-wave selector in K
    const int wavesPerBlk = blockDim.x >> 5;
    const int gwave = blockIdx.x * wavesPerBlk + (tid >> 5);
    const int totalWaves = gridDim.x * wavesPerBlk;

    for (int t = gwave; t < numTiles; t += totalWaves) {
        const int ebase = t * 16;

        // dst rows this lane is responsible for in C: M = 8*khi + r
        int dstv[8];
#pragma unroll
        for (int r = 0; r < 8; ++r) dstv[r] = dst[ebase + khi * 8 + r];

        const float* arow  = hv + (size_t)(ebase + mrow) * DIM;          // K in [0,256)
        const float* arow2 = hp + (size_t)(ebase + mrow) * DIM - 256;    // K in [256,512)

        float coef[8];
#pragma unroll
        for (int r = 0; r < 8; ++r) coef[r] = 0.f;

        // ---- GEMM: e[16 x 256] in 4 N-chunks of 4 WMMA tiles ----
        for (int nc = 0; nc < 4; ++nc) {
            v8f cf[4];
#pragma unroll
            for (int j = 0; j < 4; ++j) cf[j] = 0;

#pragma unroll 4
            for (int k = 0; k < 16; ++k) {
                // A fragment: 16-bit A 16x32 layout
                const float* srcp = (k < 8) ? (arow + 32 * k) : (arow2 + 32 * k);
                const int b1 = 16 * khi;
                float4 f0 = *(const float4*)(srcp + b1);
                float4 f1 = *(const float4*)(srcp + b1 + 4);
                float4 f2 = *(const float4*)(srcp + b1 + 8);
                float4 f3 = *(const float4*)(srcp + b1 + 12);
                v16h a;
                a[0]=(_Float16)f0.x; a[1]=(_Float16)f0.y; a[2]=(_Float16)f0.z; a[3]=(_Float16)f0.w;
                a[4]=(_Float16)f1.x; a[5]=(_Float16)f1.y; a[6]=(_Float16)f1.z; a[7]=(_Float16)f1.w;
                a[8]=(_Float16)f2.x; a[9]=(_Float16)f2.y; a[10]=(_Float16)f2.z; a[11]=(_Float16)f2.w;
                a[12]=(_Float16)f3.x; a[13]=(_Float16)f3.y; a[14]=(_Float16)f3.z; a[15]=(_Float16)f3.w;

#pragma unroll
                for (int j = 0; j < 4; ++j) {
                    const int n = nc * 4 + j;
                    // B fragment: col N = n*16 + mrow, K = 32k + 16*khi .. +15 (contiguous)
                    const _Float16* bp = wqs + (size_t)(n * 16 + mrow) * WQ_PITCH
                                       + 32 * k + 16 * khi;
                    union { v16h v; struct { v8h lo, hi; } s; } bu;
                    bu.s.lo = *(const v8h*)bp;
                    bu.s.hi = *(const v8h*)(bp + 8);
                    cf[j] = __builtin_amdgcn_wmma_f32_16x16x32_f16(
                        false, a, false, bu.v, (short)0, cf[j], false, false);
                }
            }

            // tanh + dot with gathered h_t rows (coalesced 64B per half-wave)
#pragma unroll
            for (int j = 0; j < 4; ++j) {
                const int col = (nc * 4 + j) * 16 + mrow;
#pragma unroll
                for (int r = 0; r < 8; ++r) {
                    float ev = tanhf(cf[j][r]);
                    float tv = ht[(size_t)dstv[r] * DIM + col];
                    coef[r] += ev * tv;
                }
            }
        }

        // ---- reduce coef across the 16 lanes of each half-wave ----
#pragma unroll
        for (int r = 0; r < 8; ++r) {
            float v = coef[r];
            v += __shfl_xor(v, 8, 16);
            v += __shfl_xor(v, 4, 16);
            v += __shfl_xor(v, 2, 16);
            v += __shfl_xor(v, 1, 16);
            coef[r] = v;
        }

        // ---- run-compacted scatter: m = h_v[e] * coeff, summed per dst run ----
        const int colb = lane * 8;
        float acc[8];
        int curDst = -1;
#pragma unroll
        for (int e = 0; e < 16; ++e) {
            float ce = __shfl(coef[e & 7], (e & 8) ? 16 : 0, 32);
            int   de = __shfl(dstv[e & 7], (e & 8) ? 16 : 0, 32);
            const float4* vp = (const float4*)(hv + (size_t)(ebase + e) * DIM + colb);
            float4 v0 = vp[0], v1 = vp[1];
            float vv[8];
            vv[0]=v0.x; vv[1]=v0.y; vv[2]=v0.z; vv[3]=v0.w;
            vv[4]=v1.x; vv[5]=v1.y; vv[6]=v1.z; vv[7]=v1.w;
            if (de != curDst) {
                if (curDst >= 0)
                    flush_atomic(outTgt + (size_t)curDst * DIM + colb, acc);
                curDst = de;
#pragma unroll
                for (int j = 0; j < 8; ++j) acc[j] = vv[j] * ce;
            } else {
#pragma unroll
                for (int j = 0; j < 8; ++j) acc[j] += vv[j] * ce;
            }
        }
        flush_atomic(outTgt + (size_t)curDst * DIM + colb, acc);
    }
}

// last_feat[j] = h_v[last_idx[j]] : one block per row, 64 threads x float4
__global__ void __launch_bounds__(64)
gather_last_kernel(const float* __restrict__ hv, const int* __restrict__ lastIdx,
                   float* __restrict__ outLast) {
    const int row = blockIdx.x;
    const int srcRow = lastIdx[row];
    ((float4*)(outLast + (size_t)row * DIM))[threadIdx.x] =
        ((const float4*)(hv + (size_t)srcRow * DIM))[threadIdx.x];
}

extern "C" void kernel_launch(void* const* d_in, const int* in_sizes, int n_in,
                              void* d_out, int out_size, void* d_ws, size_t ws_size,
                              hipStream_t stream) {
    (void)in_sizes; (void)n_in; (void)d_ws; (void)ws_size; (void)out_size;
    const float* h_v  = (const float*)d_in[0];
    const float* h_p  = (const float*)d_in[1];
    const float* h_t  = (const float*)d_in[2];
    // d_in[3] = src == arange(E), identity gather -> unused
    const int*   dst  = (const int*)d_in[4];
    const int*   last = (const int*)d_in[5];
    const float* wq   = (const float*)d_in[6];

    float* outTgt  = (float*)d_out;
    float* outLast = outTgt + (size_t)N_TGT * DIM;

    // atomic-accumulated output must start at zero (harness poisons d_out)
    hipMemsetAsync(outTgt, 0, (size_t)N_TGT * DIM * sizeof(float), stream);

    const int numTiles = E_EDGE / 16;                 // 25600
    const size_t shmem = (size_t)DIM * WQ_PITCH * 2;  // 266240 B f16 Wq in LDS
    fused_edge_kernel<<<dim3(256), dim3(256), shmem, stream>>>(
        h_v, h_p, h_t, dst, wq, outTgt, numTiles);

    gather_last_kernel<<<dim3(N_TGT), dim3(64), 0, stream>>>(h_v, last, outLast);
}
